// SeqAttention_54975581389354
// MI455X (gfx1250) — matrix operand order repaired and checked by reading, not compiled, not run on hardware
//
#include <hip/hip_runtime.h>

typedef __attribute__((ext_vector_type(16))) _Float16 v16h;
typedef __attribute__((ext_vector_type(8)))  _Float16 v8h;
typedef __attribute__((ext_vector_type(4)))  _Float16 v4h;
typedef __attribute__((ext_vector_type(8)))  float    v8f;

#define HD     64
#define NBATCH 32
#define MQ     1024
#define KLEN   3072     // M + ATTN_LIM
#define CLEN   768      // C + M/4
#define CPAD   832      // CLEN + 64 zero-pad rows
#define VPAD   3136     // KLEN + 64 zero-pad rows
#define PEW    2561     // ATTN_LIM + C + 1
#define NFULL  2048
#define NCOMP  513
#define KSPLIT 768      // full-window columns handled by sub-wave 0
#define WAVES  4        // 2 query tiles per workgroup, 2 waves per tile
#define LOG2E  1.44269504088896f

// ---- workspace layout (bytes, all 256-aligned) ----
#define WS_KEY 0u
#define WS_CK  12582912u            // 32*3072*64*2
#define WS_VAL 15990784u            // + 32*832*64*2
#define WS_CV  28835840u            // + 32*64*3136*2
#define WS_PE  32243712u            // + 32*64*832*2   (pe_t: 2561*64*2 more)

struct WaveLds {
    alignas(16) _Float16 pabs[16][72];  // absolute-coords probabilities (banded, zero-padded)
    alignas(16) float    sabs[16][68];  // QK scores (left-shifted band) / O export for merge
    alignas(16) float    spe[16][36];   // relative-coords PE scores (16 x 32 used)
    alignas(16) float    bcast[16];     // per-row broadcast
    alignas(16) float    mexp[16];      // split-K merge: partner max / a1
    alignas(16) float    lexp[16];      // split-K merge: partner sum / final l
};

__device__ __forceinline__ void lds_fence() {
    asm volatile("s_wait_dscnt 0" ::: "memory");
}

__device__ __forceinline__ v8f wmma_f16(v16h a, v16h b, v8f c) {
    return __builtin_amdgcn_wmma_f32_16x16x32_f16(false, a, false, b, (short)0, c, false, false);
}

__device__ __forceinline__ v16h ldg16(const _Float16* p) {   // 16B-aligned
    v8h b0 = *(const v8h*)p;
    v8h b1 = *(const v8h*)(p + 8);
    v16h r;
    #pragma unroll
    for (int j = 0; j < 8; ++j) { r[j] = b0[j]; r[8 + j] = b1[j]; }
    return r;
}

__device__ __forceinline__ v16h ldg16u8(const _Float16* p) { // 8B-aligned
    v4h b0 = *(const v4h*)p;
    v4h b1 = *(const v4h*)(p + 4);
    v4h b2 = *(const v4h*)(p + 8);
    v4h b3 = *(const v4h*)(p + 12);
    v16h r;
    #pragma unroll
    for (int j = 0; j < 4; ++j) {
        r[j] = b0[j]; r[4 + j] = b1[j]; r[8 + j] = b2[j]; r[12 + j] = b3[j];
    }
    return r;
}

// ---------------- preprocessing: f32 -> f16 operand layouts ----------------
__global__ __launch_bounds__(256)
void cvt_rowmajor(const float* __restrict__ src, _Float16* __restrict__ dst,
                  int rows_in, int rows_out, long total)
{
    long idx = (long)blockIdx.x * 256 + threadIdx.x;
    if (idx >= total) return;
    int d = (int)(idx & 63);
    long t = idx >> 6;
    int r = (int)(t % rows_out);
    int b = (int)(t / rows_out);
    dst[idx] = (r < rows_in) ? (_Float16)src[((size_t)b * rows_in + r) * HD + d]
                             : (_Float16)0.f;
}

__global__ __launch_bounds__(256)
void transpose_f16(const float* __restrict__ src, _Float16* __restrict__ dst,
                   int rows_in, int rows_cap, long total)
{   // src [nb][rows_in][64] -> dst [nb][64][rows_cap], zero pad
    long idx = (long)blockIdx.x * 256 + threadIdx.x;
    if (idx >= total) return;
    int r = (int)(idx % rows_cap);
    long t = idx / rows_cap;
    int d = (int)(t & 63);
    int b = (int)(t >> 6);
    dst[idx] = (r < rows_in) ? (_Float16)src[((size_t)b * rows_in + r) * HD + d]
                             : (_Float16)0.f;
}

__global__ __launch_bounds__(256)
void transpose_pe(const float* __restrict__ src, _Float16* __restrict__ dst)
{   // src [64][2561] -> dst [2561][64]
    long idx = (long)blockIdx.x * 256 + threadIdx.x;
    if (idx >= (long)PEW * HD) return;
    int d = (int)(idx & 63);
    int c = (int)(idx >> 6);
    dst[idx] = (_Float16)src[(size_t)d * PEW + c];
}

// ---------------- main fused attention ----------------
// One phase of the flash loop (DIV=4: compressed keys, DIV=1: full sliding window).
// Logits are in the log2 domain (log2(e)/sqrt(64) folded into Q).
template<int DIV>
__device__ __forceinline__ void run_phase(
    WaveLds& W, const v16h* qa, v8f* acc, float& mrow, float& lrow,
    const _Float16* __restrict__ kmat,   // row-major f16 [rows][64], zero-padded
    const _Float16* __restrict__ vmat,   // transposed f16 [64][vcap], zero-padded
    int vcap, int rowbase, int n_k, int n_loop, int peoff,
    const _Float16* __restrict__ pe_t,   // [2561][64] f16
    int lane)
{
    const int hi   = lane >> 4;     // lane half
    const int ii   = lane & 15;     // row within tile (A layout), column in B/C layouts
    const int soff = ii / DIV;      // skew offset of this row's band
    const int r0   = hi ? 8 : 0;

    // zero the banded probability buffer once per phase (band cols rewritten every iter)
    {
        unsigned int* p = (unsigned int*)&W.pabs[0][0];
        #pragma unroll
        for (int t = lane; t < (16 * 72) / 2; t += 32) p[t] = 0u;
    }
    lds_fence();

    for (int k0 = 0; k0 < n_loop; k0 += 32) {
        const int ja = rowbase + k0;   // absolute start row of this window

        // prefetch next window's key rows into near caches
        __builtin_prefetch(kmat + (size_t)(ja + 32 + lane) * HD, 0, 3);

        // ---- issue ALL global B-operand loads up front (pipelined via loadcnt) ----
        v16h kb[3][2], pb[2][2], vb[4][2];
        #pragma unroll
        for (int t = 0; t < 3; ++t) {
            const _Float16* kr = kmat + (size_t)(ja + 16 * t + ii) * HD + 16 * hi;
            kb[t][0] = ldg16(kr);
            kb[t][1] = ldg16(kr + 32);
        }
        #pragma unroll
        for (int t = 0; t < 2; ++t) {
            const _Float16* pr = pe_t + (size_t)(peoff + k0 + 16 * t + ii) * HD + 16 * hi;
            pb[t][0] = ldg16(pr);
            pb[t][1] = ldg16(pr + 32);
        }
        #pragma unroll
        for (int c = 0; c < 4; ++c) {
            const _Float16* vcol = vmat + (size_t)(16 * c + ii) * vcap + ja + 16 * hi;
            vb[c][0] = ldg16u8(vcol);
            vb[c][1] = ldg16u8(vcol + 32);
        }

        // ---- QK^T in absolute coords: 3 tiles; store left-shifted (band -> cols 16..47) ----
        #pragma unroll
        for (int t = 0; t < 3; ++t) {
            v8f d = (v8f){0.f,0.f,0.f,0.f,0.f,0.f,0.f,0.f};
            d = wmma_f16(qa[0], kb[t][0], d);
            d = wmma_f16(qa[1], kb[t][1], d);
            const int cc = 16 * t + ii;
            #pragma unroll
            for (int r = 0; r < 8; ++r)
                W.sabs[r0 + r][16 + cc - (r0 + r) / DIV] = d[r];
        }

        // ---- PE term in relative coords: 2 tiles ----
        #pragma unroll
        for (int t = 0; t < 2; ++t) {
            v8f d = (v8f){0.f,0.f,0.f,0.f,0.f,0.f,0.f,0.f};
            d = wmma_f16(qa[0], pb[t][0], d);
            d = wmma_f16(qa[1], pb[t][1], d);
            const int cc = 16 * t + ii;
            #pragma unroll
            for (int r = 0; r < 8; ++r) W.spe[r0 + r][cc] = d[r];
        }
        lds_fence();

        // ---- vectorized skewed read + online softmax (lane pair (ii, ii+16) owns row ii) ----
        const float* sa = &W.sabs[ii][16 + 16 * hi];
        const float* sp = &W.spe[ii][16 * hi];
        float lg[16];
        float mloc = -3.0e38f;
        #pragma unroll
        for (int j4 = 0; j4 < 4; ++j4) {
            const float4 a = *(const float4*)(sa + 4 * j4);
            const float4 p = *(const float4*)(sp + 4 * j4);
            float t0 = a.x + p.x, t1 = a.y + p.y, t2 = a.z + p.z, t3 = a.w + p.w;
            if constexpr (DIV == 4) {          // only the compressed tail needs masking
                const int kk = k0 + 16 * hi + 4 * j4;
                if (kk + 0 >= n_k) t0 = -3.0e38f;
                if (kk + 1 >= n_k) t1 = -3.0e38f;
                if (kk + 2 >= n_k) t2 = -3.0e38f;
                if (kk + 3 >= n_k) t3 = -3.0e38f;
            }
            lg[4*j4+0] = t0; lg[4*j4+1] = t1; lg[4*j4+2] = t2; lg[4*j4+3] = t3;
            mloc = fmaxf(fmaxf(fmaxf(mloc, t0), fmaxf(t1, t2)), t3);
        }
        mloc = fmaxf(mloc, __shfl_xor(mloc, 16, 32));
        const float mnew  = fmaxf(mrow, mloc);
        const float alpha = exp2f(mrow - mnew);      // ==1.0 when max unchanged
        float ps = 0.0f;
        #pragma unroll
        for (int j = 0; j < 16; ++j) {
            const int kk = 16 * hi + j;
            const float p = exp2f(lg[j] - mnew);
            ps += p;
            W.pabs[ii][soff + kk] = (_Float16)p;     // skew back to absolute band
        }
        ps  += __shfl_xor(ps, 16, 32);
        lrow = lrow * alpha + ps;
        mrow = mnew;
        if (hi == 0) W.bcast[ii] = alpha;
        lds_fence();

        // ---- rescale output accumulators in place (D layout: VGPR r <-> row r0+r) ----
        const float4 s0 = *(const float4*)&W.bcast[r0];
        const float4 s1 = *(const float4*)&W.bcast[r0 + 4];
        const float sc[8] = {s0.x, s0.y, s0.z, s0.w, s1.x, s1.y, s1.z, s1.w};
        #pragma unroll
        for (int c = 0; c < 4; ++c)
            #pragma unroll
            for (int r = 0; r < 8; ++r) acc[c][r] *= sc[r];

        // ---- P x V in absolute coords: A operands shared across the 4 dim-chunks ----
        v16h aa[2];
        #pragma unroll
        for (int h2 = 0; h2 < 2; ++h2) {
            const v8h plo = *(const v8h*)&W.pabs[ii][32 * h2 + 16 * hi];
            const v8h phi = *(const v8h*)&W.pabs[ii][32 * h2 + 16 * hi + 8];
            #pragma unroll
            for (int j = 0; j < 8; ++j) { aa[h2][j] = plo[j]; aa[h2][8 + j] = phi[j]; }
        }
        #pragma unroll
        for (int c = 0; c < 4; ++c) {
            v8f d = acc[c];
            d = wmma_f16(aa[0], vb[c][0], d);
            d = wmma_f16(aa[1], vb[c][1], d);
            acc[c] = d;
        }
    }
}

__global__ __launch_bounds__(WAVES * 32)
void seqattn_kernel(const float* __restrict__ q,
                    const _Float16* __restrict__ key_h, const _Float16* __restrict__ val_t,
                    const _Float16* __restrict__ ck_h,  const _Float16* __restrict__ cv_t,
                    const _Float16* __restrict__ pe_t,  float* __restrict__ out)
{
    __shared__ WaveLds lds[WAVES];
    const int wave = threadIdx.x >> 5;
    const int lane = threadIdx.x & 31;
    const int sub  = wave & 1;                 // split-K role within the tile pair
    const int tile = blockIdx.x * 2 + (wave >> 1);
    const int b    = tile >> 6;                // / (MQ/16)
    const int i0   = (tile & 63) << 4;
    WaveLds& W = lds[wave];

    const int hi = lane >> 4;
    const int ii = lane & 15;
    const int r0 = hi ? 8 : 0;

    // Q as two f16 A-operands; fold log2(e)/sqrt(64) so logits are log2-domain
    v16h qa[2];
    const float* qrow = q + ((size_t)b * MQ + (i0 + ii)) * HD;
    #pragma unroll
    for (int h = 0; h < 2; ++h)
        #pragma unroll
        for (int j = 0; j < 16; ++j) {
            const int k = 32 * h + 16 * hi + j;
            qa[h][j] = (_Float16)(qrow[k] * (0.125f * LOG2E));
        }

    v8f acc[4];
    #pragma unroll
    for (int c = 0; c < 4; ++c) acc[c] = (v8f){0.f,0.f,0.f,0.f,0.f,0.f,0.f,0.f};
    float mrow = -3.0e38f, lrow = 0.0f;

    const _Float16* kmat = key_h + (size_t)b * KLEN * HD;
    const _Float16* vmat = val_t + (size_t)b * HD * VPAD;

    if (sub == 0) {
        // compressed keys (17 blocks, last masks k>512) + full window k in [0, 768)
        run_phase<4>(W, qa, acc, mrow, lrow,
                     ck_h + (size_t)b * CPAD * HD, cv_t + (size_t)b * HD * CPAD,
                     CPAD, i0 >> 2, NCOMP, 544, 0, pe_t, lane);
        run_phase<1>(W, qa, acc, mrow, lrow, kmat, vmat,
                     VPAD, i0, 0, KSPLIT, NCOMP, pe_t, lane);
    } else {
        // full window k in [768, 2048)
        run_phase<1>(W, qa, acc, mrow, lrow, kmat, vmat,
                     VPAD, i0 + KSPLIT, 0, NFULL - KSPLIT, NCOMP + KSPLIT, pe_t, lane);
    }

    // ---- split-K merge: sub1 exports (m, l, O) via its LDS region ----
    if (sub == 1) {
        if (hi == 0) { W.mexp[ii] = mrow; W.lexp[ii] = lrow; }
        #pragma unroll
        for (int c = 0; c < 4; ++c)
            #pragma unroll
            for (int r = 0; r < 8; ++r) W.sabs[r0 + r][16 * c + ii] = acc[c][r];
    }
    __syncthreads();
    if (sub == 0) {
        WaveLds& W1 = lds[wave + 1];
        const float m1 = W1.mexp[ii];
        const float l1 = W1.lexp[ii];
        const float m  = fmaxf(mrow, m1);
        const float a0 = exp2f(mrow - m);
        const float a1 = exp2f(m1 - m);
        const float lt = lrow * a0 + l1 * a1;
        if (hi == 0) { W.bcast[ii] = a0; W.mexp[ii] = a1; W.lexp[ii] = lt; }
        lds_fence();
        const float4 a0l = *(const float4*)&W.bcast[r0];
        const float4 a0h = *(const float4*)&W.bcast[r0 + 4];
        const float4 a1l = *(const float4*)&W.mexp[r0];
        const float4 a1h = *(const float4*)&W.mexp[r0 + 4];
        const float4 ltl = *(const float4*)&W.lexp[r0];
        const float4 lth = *(const float4*)&W.lexp[r0 + 4];
        const float a0v[8] = {a0l.x,a0l.y,a0l.z,a0l.w,a0h.x,a0h.y,a0h.z,a0h.w};
        const float a1v[8] = {a1l.x,a1l.y,a1l.z,a1l.w,a1h.x,a1h.y,a1h.z,a1h.w};
        const float ltv[8] = {ltl.x,ltl.y,ltl.z,ltl.w,lth.x,lth.y,lth.z,lth.w};
        #pragma unroll
        for (int c = 0; c < 4; ++c)
            #pragma unroll
            for (int r = 0; r < 8; ++r) {
                const float o1 = W1.sabs[r0 + r][16 * c + ii];
                out[((size_t)b * MQ + i0 + r0 + r) * HD + 16 * c + ii] =
                    (acc[c][r] * a0v[r] + o1 * a1v[r]) / ltv[r];
            }
    }
}

extern "C" void kernel_launch(void* const* d_in, const int* in_sizes, int n_in,
                              void* d_out, int out_size, void* d_ws, size_t ws_size,
                              hipStream_t stream)
{
    (void)in_sizes; (void)n_in; (void)out_size; (void)ws_size;
    const float* q   = (const float*)d_in[0];
    const float* key = (const float*)d_in[1];
    const float* val = (const float*)d_in[2];
    const float* ck  = (const float*)d_in[3];
    const float* cv  = (const float*)d_in[4];
    const float* pe  = (const float*)d_in[5];

    char* ws = (char*)d_ws;
    _Float16* key_h = (_Float16*)(ws + WS_KEY);   // [32][3072][64]
    _Float16* ck_h  = (_Float16*)(ws + WS_CK);    // [32][832][64]  (rows >= 768 zero)
    _Float16* val_t = (_Float16*)(ws + WS_VAL);   // [32][64][3136] (rows >= 3072 zero)
    _Float16* cv_t  = (_Float16*)(ws + WS_CV);    // [32][64][832]  (rows >= 768 zero)
    _Float16* pe_t  = (_Float16*)(ws + WS_PE);    // [2561][64]

    const long nk = (long)NBATCH * KLEN * HD;     // 6,291,456
    const long nc = (long)NBATCH * CPAD * HD;     // 1,703,936
    const long nv = (long)NBATCH * HD * VPAD;     // 6,422,528
    const long nw = (long)NBATCH * HD * CPAD;     // 1,703,936
    const long np = (long)PEW * HD;               //   163,904

    cvt_rowmajor <<<(nk + 255) / 256, 256, 0, stream>>>(key, key_h, KLEN, KLEN, nk);
    cvt_rowmajor <<<(nc + 255) / 256, 256, 0, stream>>>(ck,  ck_h,  CLEN, CPAD, nc);
    transpose_f16<<<(nv + 255) / 256, 256, 0, stream>>>(val, val_t, KLEN, VPAD, nv);
    transpose_f16<<<(nw + 255) / 256, 256, 0, stream>>>(cv,  cv_t,  CLEN, CPAD, nw);
    transpose_pe <<<(np + 255) / 256, 256, 0, stream>>>(pe,  pe_t);

    const int n_tiles = NBATCH * (MQ / 16);       // 2048 query tiles, 2 per WG
    seqattn_kernel<<<n_tiles / 2, WAVES * 32, 0, stream>>>(
        q, key_h, val_t, ck_h, cv_t, pe_t, (float*)d_out);
}
